// StateSequencePredictor_52939766890869
// MI455X (gfx1250) — matrix operand order repaired and checked by reading, not compiled, not run on hardware
//
#include <hip/hip_runtime.h>
#include <hip/hip_bf16.h>

// ---------------------------------------------------------------------------
// Types
// ---------------------------------------------------------------------------
typedef __attribute__((ext_vector_type(16))) __bf16 v16bf;
typedef __attribute__((ext_vector_type(8)))  __bf16 v8bf;
typedef __attribute__((ext_vector_type(8)))  float  v8f;

#define B_TOT   16384
#define EMBED   512
#define STATE   128
#define HIDDEN  256
#define HORIZON 32

#define RS_H 264   // padded LDS row stride (bf16 elems) for 256-wide h
#define RS_S 136   // padded LDS row stride (bf16 elems) for 128-wide state

// ---------------------------------------------------------------------------
// WMMA helper: D = A(16x32 bf16) * B(32x16 bf16) + C(16x16 f32)
// ---------------------------------------------------------------------------
__device__ __forceinline__ v8f wmma_bf16(v16bf a, v16bf b, v8f c) {
    return __builtin_amdgcn_wmma_f32_16x16x32_bf16(
        /*neg_a=*/false, a, /*neg_b=*/false, b,
        /*c_mod=*/(short)0, c, /*reuse_a=*/false, /*reuse_b=*/false);
}

// A-operand fragment from a row-major bf16 matrix (16 rows, stride `stride`)
// lane l: row = l&15, K chunks [c*8, c*8+8) and [16+c*8, ...), c = l>>4
__device__ __forceinline__ v16bf load_a(const __bf16* base, int stride, int kt, int lane) {
    int row = lane & 15;
    int c8  = (lane >> 4) << 3;            // 0 or 8
    const __bf16* p = base + row * stride + kt * 32 + c8;
    v8bf lo = *(const v8bf*)(p);
    v8bf hi = *(const v8bf*)(p + 16);
    v16bf a;
#pragma unroll
    for (int i = 0; i < 8; ++i) { a[i] = lo[i]; a[i + 8] = hi[i]; }
    return a;
}

// B-operand fragment from pre-packed weights: one contiguous 32B load per lane
__device__ __forceinline__ v16bf load_b(const __bf16* pk, int KT, int nt, int kt, int lane) {
    return *(const v16bf*)(pk + (((size_t)nt * KT + kt) * 32 + lane) * 16);
}

__device__ __forceinline__ float sigmoidf_(float x) {
    return 1.0f / (1.0f + __expf(-x));
}

// ---------------------------------------------------------------------------
// Kernel 1: pack W [N x K] (row-major f32, used as B = W^T) into per-lane
// bf16 WMMA B-fragment order: frag(nt,kt), lane l -> W[nt*16+(l&15)]
//                 [kt*32+(l>>4)*8 .. +8) then [kt*32+16+(l>>4)*8 .. +8)
// ---------------------------------------------------------------------------
__global__ void pack_b_kernel(const float* __restrict__ W, __bf16* __restrict__ dst,
                              int N, int K) {
    int t = blockIdx.x * blockDim.x + threadIdx.x;
    int KT = K >> 5;
    int total = (N >> 4) * KT * 32;
    if (t >= total) return;
    int lane = t & 31;
    int frag = t >> 5;
    int nt = frag / KT;
    int kt = frag - nt * KT;
    int n  = nt * 16 + (lane & 15);
    int kb = kt * 32 + ((lane >> 4) << 3);
    const float* src = W + (size_t)n * K + kb;
    __bf16* d = dst + (size_t)t * 16;
#pragma unroll
    for (int e = 0; e < 8; ++e) { d[e] = (__bf16)src[e]; d[e + 8] = (__bf16)src[16 + e]; }
}

// ---------------------------------------------------------------------------
// Kernel 2: h_init = embedding @ Wp^T + bp   ([16384,512] x [512,256])
// block = 128 threads (4 waves); wave w -> 64-col strip; grid.x = 1024 m-tiles
// ---------------------------------------------------------------------------
__global__ void __launch_bounds__(128)
hinit_kernel(const float* __restrict__ emb, const __bf16* __restrict__ Wp_pk,
             const float* __restrict__ bp, float* __restrict__ hinit) {
    int lane = threadIdx.x & 31;
    int wave = threadIdx.x >> 5;          // 0..3 column strip
    int mt   = blockIdx.x;                // 0..1023
    int row  = lane & 15;
    int c8   = (lane >> 4) << 3;
    const float* arow = emb + (size_t)(mt * 16 + row) * EMBED;

    v8f acc[4] = {};
#pragma unroll 1
    for (int kt = 0; kt < EMBED / 32; ++kt) {
        v8f x0 = *(const v8f*)(arow + kt * 32 + c8);
        v8f x1 = *(const v8f*)(arow + kt * 32 + 16 + c8);
        v16bf a;
#pragma unroll
        for (int i = 0; i < 8; ++i) { a[i] = (__bf16)x0[i]; a[i + 8] = (__bf16)x1[i]; }
#pragma unroll
        for (int tt = 0; tt < 4; ++tt)
            acc[tt] = wmma_bf16(a, load_b(Wp_pk, EMBED / 32, wave * 4 + tt, kt, lane), acc[tt]);
    }

    int col = lane & 15;
    int rb  = (lane >> 4) * 8;
#pragma unroll
    for (int tt = 0; tt < 4; ++tt) {
        int n0 = wave * 64 + tt * 16 + col;
        float b = bp[n0];
#pragma unroll
        for (int v = 0; v < 8; ++v) {
            int m = mt * 16 + rb + v;
            hinit[(size_t)m * HIDDEN + n0] = acc[tt][v] + b;
        }
    }
}

// ---------------------------------------------------------------------------
// GRU cell for a 16-row tile: x [16,K=KTX*32] in LDS, h [16,256] in LDS.
// A-fragments are hoisted into registers once per cell (invariant over jt);
// the jt loop is then pure weight-stream (L2) + WMMA on 6 accumulators.
// Writes h_new (bf16) into hn. Gate order r,z,n (PyTorch).
// ---------------------------------------------------------------------------
template <int KTX, int XSTR>
__device__ __forceinline__ void gru_cell(const __bf16* __restrict__ xs,
                                         const __bf16* __restrict__ hc,
                                         __bf16* __restrict__ hn,
                                         const __bf16* __restrict__ Wih,
                                         const __bf16* __restrict__ Whh,
                                         const float* __restrict__ bih,
                                         const float* __restrict__ bhh, int lane) {
    int col = lane & 15;
    int rb  = (lane >> 4) * 8;

    // Hoisted A fragments (stay resident in VGPRs across the whole cell)
    v16bf ax[KTX];
#pragma unroll
    for (int kt = 0; kt < KTX; ++kt) ax[kt] = load_a(xs, XSTR, kt, lane);
    v16bf ah[HIDDEN / 32];
#pragma unroll
    for (int kt = 0; kt < HIDDEN / 32; ++kt) ah[kt] = load_a(hc, RS_H, kt, lane);

#pragma unroll 1
    for (int jt = 0; jt < HIDDEN / 16; ++jt) {
        v8f air = {}, aiz = {}, ain = {}, ahr = {}, ahz = {}, ahn = {};
#pragma unroll
        for (int kt = 0; kt < KTX; ++kt) {
            air = wmma_bf16(ax[kt], load_b(Wih, KTX, jt,      kt, lane), air);
            aiz = wmma_bf16(ax[kt], load_b(Wih, KTX, 16 + jt, kt, lane), aiz);
            ain = wmma_bf16(ax[kt], load_b(Wih, KTX, 32 + jt, kt, lane), ain);
        }
#pragma unroll
        for (int kt = 0; kt < HIDDEN / 32; ++kt) {
            ahr = wmma_bf16(ah[kt], load_b(Whh, HIDDEN / 32, jt,      kt, lane), ahr);
            ahz = wmma_bf16(ah[kt], load_b(Whh, HIDDEN / 32, 16 + jt, kt, lane), ahz);
            ahn = wmma_bf16(ah[kt], load_b(Whh, HIDDEN / 32, 32 + jt, kt, lane), ahn);
        }
        int n0 = jt * 16 + col;
        float bir = bih[n0], biz = bih[256 + n0], bi_n = bih[512 + n0];
        float bhr = bhh[n0], bhz = bhh[256 + n0], bh_n = bhh[512 + n0];
#pragma unroll
        for (int v = 0; v < 8; ++v) {
            int m   = rb + v;
            float r = sigmoidf_(air[v] + bir + ahr[v] + bhr);
            float z = sigmoidf_(aiz[v] + biz + ahz[v] + bhz);
            float nn = tanhf(ain[v] + bi_n + r * (ahn[v] + bh_n));
            float hold = (float)hc[m * RS_H + n0];
            hn[m * RS_H + n0] = (__bf16)((1.0f - z) * nn + z * hold);
        }
    }
}

// ---------------------------------------------------------------------------
// Kernel 3: full 32-step recurrence. One wave (32 threads) per 16 batch rows.
// All recurrent state lives in LDS (bf16, padded, double-buffered h).
// ---------------------------------------------------------------------------
__global__ void __launch_bounds__(32)
gru_kernel(const float* __restrict__ state0, const float* __restrict__ hinit,
           const __bf16* __restrict__ Wih0, const __bf16* __restrict__ Whh0,
           const __bf16* __restrict__ Wih1, const __bf16* __restrict__ Whh1,
           const __bf16* __restrict__ Wo,
           const float* __restrict__ bih0, const float* __restrict__ bhh0,
           const float* __restrict__ bih1, const float* __restrict__ bhh1,
           const float* __restrict__ bo, float* __restrict__ out) {
    __shared__ alignas(16) __bf16 s_state[16 * RS_S];
    __shared__ alignas(16) __bf16 s_h0[2][16 * RS_H];
    __shared__ alignas(16) __bf16 s_h1[2][16 * RS_H];

    int lane  = threadIdx.x;
    size_t m0 = (size_t)blockIdx.x * 16;

    // init: h0 = h1 = h_init rows; state = initial_state rows
    for (int idx = lane; idx < 16 * HIDDEN; idx += 32) {
        int r = idx >> 8, c = idx & 255;
        __bf16 h = (__bf16)hinit[(m0 + r) * HIDDEN + c];
        s_h0[0][r * RS_H + c] = h;
        s_h1[0][r * RS_H + c] = h;
    }
    for (int idx = lane; idx < 16 * STATE; idx += 32) {
        int r = idx >> 7, c = idx & 127;
        s_state[r * RS_S + c] = (__bf16)state0[(m0 + r) * STATE + c];
    }
    asm volatile("s_wait_dscnt 0" ::: "memory");

    int cur0 = 0, cur1 = 0;
    int col = lane & 15;
    int rb  = (lane >> 4) * 8;

    for (int t = 0; t < HORIZON; ++t) {
        gru_cell<STATE / 32, RS_S>(s_state, s_h0[cur0], s_h0[cur0 ^ 1],
                                   Wih0, Whh0, bih0, bhh0, lane);
        asm volatile("s_wait_dscnt 0" ::: "memory");
        cur0 ^= 1;

        gru_cell<HIDDEN / 32, RS_H>(s_h0[cur0], s_h1[cur1], s_h1[cur1 ^ 1],
                                    Wih1, Whh1, bih1, bhh1, lane);
        asm volatile("s_wait_dscnt 0" ::: "memory");
        cur1 ^= 1;

        // next_state = h1 @ Wo^T + bo  -> d_out and LDS state (bf16)
        v16bf a1[HIDDEN / 32];
#pragma unroll
        for (int kt = 0; kt < HIDDEN / 32; ++kt)
            a1[kt] = load_a(s_h1[cur1], RS_H, kt, lane);

#pragma unroll 1
        for (int jt = 0; jt < STATE / 16; ++jt) {
            v8f acc = {};
#pragma unroll
            for (int kt = 0; kt < HIDDEN / 32; ++kt)
                acc = wmma_bf16(a1[kt], load_b(Wo, HIDDEN / 32, jt, kt, lane), acc);
            int n0  = jt * 16 + col;
            float b = bo[n0];
#pragma unroll
            for (int v = 0; v < 8; ++v) {
                int m   = rb + v;
                float s = acc[v] + b;
                out[((m0 + m) * HORIZON + t) * STATE + n0] = s;
                s_state[m * RS_S + n0] = (__bf16)s;
            }
        }
        asm volatile("s_wait_dscnt 0" ::: "memory");
    }
}

// ---------------------------------------------------------------------------
// Host launch
// ---------------------------------------------------------------------------
extern "C" void kernel_launch(void* const* d_in, const int* in_sizes, int n_in,
                              void* d_out, int out_size, void* d_ws, size_t ws_size,
                              hipStream_t stream) {
    const float* emb   = (const float*)d_in[0];
    const float* st0   = (const float*)d_in[1];
    const float* Wp    = (const float*)d_in[2];
    const float* bp    = (const float*)d_in[3];
    const float* Wih0  = (const float*)d_in[4];
    const float* Whh0  = (const float*)d_in[5];
    const float* bih0  = (const float*)d_in[6];
    const float* bhh0  = (const float*)d_in[7];
    const float* Wih1  = (const float*)d_in[8];
    const float* Whh1  = (const float*)d_in[9];
    const float* bih1  = (const float*)d_in[10];
    const float* bhh1  = (const float*)d_in[11];
    const float* Wo    = (const float*)d_in[12];
    const float* bo    = (const float*)d_in[13];
    float* out = (float*)d_out;

    // workspace layout (bf16 elements)
    __bf16* ws = (__bf16*)d_ws;
    size_t o = 0;
    __bf16* Wp_pk   = ws + o; o += (256 / 16) * (512 / 32) * 512;   // 131072
    __bf16* Wih0_pk = ws + o; o += (768 / 16) * (128 / 32) * 512;   //  98304
    __bf16* Whh0_pk = ws + o; o += (768 / 16) * (256 / 32) * 512;   // 196608
    __bf16* Wih1_pk = ws + o; o += (768 / 16) * (256 / 32) * 512;
    __bf16* Whh1_pk = ws + o; o += (768 / 16) * (256 / 32) * 512;
    __bf16* Wo_pk   = ws + o; o += (128 / 16) * (256 / 32) * 512;   //  32768
    float* h_init = (float*)(ws + o);                               // 16384*256 f32

    auto pack = [&](const float* W, __bf16* dst, int N, int K) {
        int total = (N / 16) * (K / 32) * 32;
        pack_b_kernel<<<dim3((total + 255) / 256), dim3(256), 0, stream>>>(W, dst, N, K);
    };
    pack(Wp,   Wp_pk,   256, 512);
    pack(Wih0, Wih0_pk, 768, 128);
    pack(Whh0, Whh0_pk, 768, 256);
    pack(Wih1, Wih1_pk, 768, 256);
    pack(Whh1, Whh1_pk, 768, 256);
    pack(Wo,   Wo_pk,   128, 256);

    hinit_kernel<<<dim3(B_TOT / 16), dim3(128), 0, stream>>>(emb, Wp_pk, bp, h_init);

    gru_kernel<<<dim3(B_TOT / 16), dim3(32), 0, stream>>>(
        st0, h_init, Wih0_pk, Whh0_pk, Wih1_pk, Whh1_pk, Wo_pk,
        bih0, bhh0, bih1, bhh1, bo, out);
}
